// MultiheadAttention_56727928046401
// MI455X (gfx1250) — matrix operand order repaired and checked by reading, not compiled, not run on hardware
//
#include <hip/hip_runtime.h>

// Problem constants
#define T_LEN 1024
#define BSZ   4
#define EMB   1024
#define NH    16
#define DHD   64
#define NROWS (T_LEN * BSZ)   // 4096
#define NBIAS 2047            // deltas s - t in [-1023, 1023]

// Padded LDS row lengths (halfs) -> conflict-free B-operand reads
#define KROW 72               // 64 + 8  (144 B row stride, 36 dwords)
#define VROW 40               // 32 + 8  ( 80 B row stride, 20 dwords)

typedef __attribute__((ext_vector_type(16))) _Float16 v16h;
typedef __attribute__((ext_vector_type(8)))  _Float16 v8h;
typedef __attribute__((ext_vector_type(4)))  _Float16 v4h;
typedef __attribute__((ext_vector_type(8)))  float    v8f;

__device__ __forceinline__ v16h cat16(v8h lo, v8h hi) {
    return __builtin_shufflevector(lo, hi, 0,1,2,3,4,5,6,7,8,9,10,11,12,13,14,15);
}
__device__ __forceinline__ v8f wmma_f16(v16h a, v16h b, v8f c) {
    // D = A(16x32 f16) * B(32x16 f16) + C(16x16 f32)
    return __builtin_amdgcn_wmma_f32_16x16x32_f16(false, a, false, b, (short)0, c, false, false);
}
// Async 16-byte copy global -> LDS (per-lane addresses), tracked by ASYNCcnt.
__device__ __forceinline__ void async_copy_b128(unsigned int lds_off, const void* gptr) {
    asm volatile("global_load_async_to_lds_b128 %0, %1, off"
                 :: "v"(lds_off), "v"((unsigned long long)(uintptr_t)gptr)
                 : "memory");
}

// ---------------------------------------------------------------------------
// f32 -> f16 elementwise conversion (vectorized x4; all sizes divisible by 4)
// ---------------------------------------------------------------------------
__global__ void cvt_f32_to_f16(const float* __restrict__ src, _Float16* __restrict__ dst, int nelem) {
    int i = (blockIdx.x * blockDim.x + threadIdx.x) * 4;
    if (i < nelem) {
        float4 v = *(const float4*)(src + i);
        v4h o;
        o[0] = (_Float16)v.x; o[1] = (_Float16)v.y; o[2] = (_Float16)v.z; o[3] = (_Float16)v.w;
        *(v4h*)(dst + i) = o;
    }
}

// ---------------------------------------------------------------------------
// Gate kernel: gate_a_1[b,h,t] = sigmoid(sum l0..3) * (sigmoid(sum l4..7)*grep_a[h] - 1) + 2
// ---------------------------------------------------------------------------
__global__ void gates_kernel(const float* __restrict__ query, const float* __restrict__ gw,
                             const float* __restrict__ gb, const float* __restrict__ ga,
                             float* __restrict__ gates) {
    int idx = blockIdx.x * 256 + threadIdx.x;           // (b*NH + h)*T + t
    int t = idx & (T_LEN - 1);
    int h = (idx >> 10) & (NH - 1);
    int b = idx >> 14;
    const float* qp = query + ((size_t)(t * BSZ + b) << 10) + (h << 6);
    float s[8];
#pragma unroll
    for (int o = 0; o < 8; ++o) s[o] = gb[o];
    for (int d = 0; d < DHD; ++d) {
        float qv = qp[d];
#pragma unroll
        for (int o = 0; o < 8; ++o) s[o] += qv * gw[o * DHD + d];
    }
    float la = s[0] + s[1] + s[2] + s[3];
    float lb = s[4] + s[5] + s[6] + s[7];
    float A  = 1.0f / (1.0f + __expf(-la));
    float Bv = 1.0f / (1.0f + __expf(-lb));
    gates[idx] = A * (Bv * ga[h] - 1.0f) + 2.0f;
}

// ---------------------------------------------------------------------------
// Relative-position bias table: br[h][delta] = rel_bias[bucket(delta)][h]
// ---------------------------------------------------------------------------
__global__ void biasrow_kernel(const float* __restrict__ rel_bias, float* __restrict__ br) {
    int i = blockIdx.x * 256 + threadIdx.x;
    if (i >= NBIAS) return;
    int rp   = i - 1023;                 // rp = s - t
    int base = (rp > 0) ? 16 : 0;
    int arp  = (rp < 0) ? -rp : rp;
    int bkt;
    if (arp < 8) {
        bkt = base + arp;
    } else {
        int large = 8 + (int)(__logf((float)arp * 0.125f) * (8.0f / __logf(16.0f)));
        if (large > 15) large = 15;
        bkt = base + large;
    }
#pragma unroll
    for (int h = 0; h < NH; ++h) br[h * NBIAS + i] = rel_bias[bkt * NH + h];
}

// ---------------------------------------------------------------------------
// WMMA GEMM: Y = X(4096x1024 f16) * W^T + bias.
// Block tile 128x128 (8 waves as 4 M x 2 N); wave tile 32x64 (8 accumulators).
// Register double-buffering: loads for K-step k+1 issue before WMMAs of k.
// Epilogue modes: 0 Q->[b][h][t][d]*scale, 1 K->[b][h][t][d],
//                 2 V->[b][h][d][t] (transposed), 3 f32 d_out.
// ---------------------------------------------------------------------------
__global__ __launch_bounds__(256) void gemm16(const _Float16* __restrict__ X,
                                              const _Float16* __restrict__ W,
                                              const float* __restrict__ bias,
                                              void* __restrict__ dst, int mode, float scale) {
    const int lane = threadIdx.x & 31;
    const int wave = threadIdx.x >> 5;
    const int n    = lane & 15;
    const int hi   = lane >> 4;
    const int wm   = wave & 3;
    const int wn   = wave >> 2;
    const int rowbase = blockIdx.x * 128 + wm * 32;
    const int colbase = blockIdx.y * 128 + wn * 64;

    v8f acc[2][4] = {};
    const _Float16* xrow0 = X + (size_t)(rowbase + n) * EMB;
    const _Float16* xrow1 = xrow0 + (size_t)16 * EMB;
    const _Float16* wbase = W + (size_t)(colbase + n) * EMB + 16 * hi;

    // Prime K-step 0
    v16h a0 = *(const v16h*)(xrow0 + 16 * hi);
    v16h a1 = *(const v16h*)(xrow1 + 16 * hi);
    v16h bm[4];
#pragma unroll
    for (int nt = 0; nt < 4; ++nt) bm[nt] = *(const v16h*)(wbase + (size_t)nt * 16 * EMB);

    for (int k0 = 0; k0 < EMB; k0 += 32) {
        v16h na0 = a0, na1 = a1;
        v16h nb0 = bm[0], nb1 = bm[1], nb2 = bm[2], nb3 = bm[3];
        const int k1 = k0 + 32;
        if (k1 < EMB) {   // prefetch next K-step into registers (overlaps WMMAs)
            na0 = *(const v16h*)(xrow0 + k1 + 16 * hi);
            na1 = *(const v16h*)(xrow1 + k1 + 16 * hi);
            nb0 = *(const v16h*)(wbase + (size_t)0 * 16 * EMB + k1);
            nb1 = *(const v16h*)(wbase + (size_t)1 * 16 * EMB + k1);
            nb2 = *(const v16h*)(wbase + (size_t)2 * 16 * EMB + k1);
            nb3 = *(const v16h*)(wbase + (size_t)3 * 16 * EMB + k1);
        }
#pragma unroll
        for (int nt = 0; nt < 4; ++nt) {
            acc[0][nt] = wmma_f16(a0, bm[nt], acc[0][nt]);
            acc[1][nt] = wmma_f16(a1, bm[nt], acc[1][nt]);
        }
        a0 = na0; a1 = na1;
        bm[0] = nb0; bm[1] = nb1; bm[2] = nb2; bm[3] = nb3;
    }

#pragma unroll
    for (int mt = 0; mt < 2; ++mt)
#pragma unroll
        for (int nt = 0; nt < 4; ++nt) {
            int j = colbase + nt * 16 + n;
            float bj = bias[j];
            int h = j >> 6, d = j & 63;
#pragma unroll
            for (int r = 0; r < 8; ++r) {
                int i = rowbase + mt * 16 + 8 * hi + r;   // D layout: M = r + 8*hi
                float val = (acc[mt][nt][r] + bj) * scale;
                int t = i >> 2, b = i & 3;                // row = t*BSZ + b
                if (mode == 0 || mode == 1) {
                    _Float16* out = (_Float16*)dst;
                    out[(((size_t)(b * NH + h)) * T_LEN + t) * DHD + d] = (_Float16)val;
                } else if (mode == 2) {
                    _Float16* out = (_Float16*)dst;
                    out[(((size_t)(b * NH + h)) * DHD + d) * T_LEN + t] = (_Float16)val;
                } else {
                    float* out = (float*)dst;
                    out[(size_t)i * EMB + j] = val;
                }
            }
        }
}

// ---------------------------------------------------------------------------
// Flash attention per (b,h): 8 waves x 16 query rows = 128 queries per block.
// K/V tiles for each 32-key block are staged cooperatively into LDS with
// global_load_async_to_lds_b128 (ASYNCcnt), double-buffered: the copy of
// block i+1 overlaps the softmax+WMMA compute of block i.
// ---------------------------------------------------------------------------
__global__ __launch_bounds__(256) void attn_kernel(const _Float16* __restrict__ Q,
                                                   const _Float16* __restrict__ K,
                                                   const _Float16* __restrict__ Vt,
                                                   const float* __restrict__ gates,
                                                   const float* __restrict__ biasrows,
                                                   _Float16* __restrict__ Ctx) {
    __shared__ __align__(16) float    sbias[NBIAS + 1];
    __shared__ __align__(16) _Float16 kl[2][32][KROW];   // K tile: [s-local][d], padded
    __shared__ __align__(16) _Float16 vl[2][64][VROW];   // V tile: [d][s-local], padded
    __shared__ __align__(16) _Float16 pl[8][16][32];     // per-wave P transpose buffer

    const int qc = blockIdx.x, h = blockIdx.y, b = blockIdx.z;
    const int tid  = threadIdx.x;
    const int lane = tid & 31;
    const int wave = tid >> 5;
    const int n  = lane & 15;
    const int hi = lane >> 4;
    const int t0 = qc * 128 + wave * 16;
    const size_t bh = (size_t)(b * NH + h);

    const _Float16* Kbh = K  + bh * (size_t)T_LEN * DHD;
    const _Float16* Vbh = Vt + bh * (size_t)DHD * T_LEN;

    // Cooperative async-copy assignments: 256 threads x 16B cover each 4KB tile.
    const int krow = tid >> 3, kcol = (tid & 7) * 8;   // K: 32 rows x 64 halfs
    const int vrow = tid >> 2, vcol = (tid & 3) * 8;   // V: 64 rows x 32 halfs
    auto issue_tiles = [&](int buf, int s0) {
        async_copy_b128((unsigned int)(uintptr_t)&kl[buf][krow][kcol],
                        Kbh + (size_t)(s0 + krow) * DHD + kcol);
        async_copy_b128((unsigned int)(uintptr_t)&vl[buf][vrow][vcol],
                        Vbh + (size_t)vrow * T_LEN + s0 + vcol);
    };

    issue_tiles(0, 0);   // prologue: stage first K/V block

    // Stage this head's bias row (8KB) with normal loads (overlaps async copy)
    for (int i = tid; i < NBIAS; i += 256) sbias[i] = biasrows[h * NBIAS + i];

    // Q tile in A-layout (rows t0..t0+15, K-dim = d): two 32-wide K-steps
    const _Float16* Qp = Q + (bh * T_LEN + t0 + n) * DHD;
    v16h aq0 = *(const v16h*)(Qp + 16 * hi);
    v16h aq1 = *(const v16h*)(Qp + 32 + 16 * hi);

    float gr[8];
#pragma unroll
    for (int r = 0; r < 8; ++r) gr[r] = gates[bh * T_LEN + t0 + 8 * hi + r];

    float mst[8], lst[8];
#pragma unroll
    for (int r = 0; r < 8; ++r) { mst[r] = -1e30f; lst[r] = 0.0f; }
    v8f o[4] = {};

    int cb = 0;
    for (int s0 = 0; s0 < T_LEN; s0 += 32) {
        if (s0 + 32 < T_LEN) {
            issue_tiles(cb ^ 1, s0 + 32);                       // overlap next copy
            asm volatile("s_wait_asynccnt 0x2" ::: "memory");   // current buffer done
        } else {
            asm volatile("s_wait_asynccnt 0x0" ::: "memory");
        }
        __syncthreads();   // all waves' copies of buffer cb are in LDS

        // S tiles from LDS K (B-layout: row = s-local, cols = d)
        v8f S[2] = {};
#pragma unroll
        for (int c = 0; c < 2; ++c) {
            const _Float16* kr = &kl[cb][c * 16 + n][0];
            S[c] = wmma_f16(aq0, *(const v16h*)(kr + 16 * hi), S[c]);
            S[c] = wmma_f16(aq1, *(const v16h*)(kr + 32 + 16 * hi), S[c]);
        }

        // gated bias + online softmax; rows r+8*hi live in 16-lane half `hi`
        float pnew[2][8];
#pragma unroll
        for (int r = 0; r < 8; ++r) {
            int t = t0 + 8 * hi + r;
            float v0 = S[0][r] + gr[r] * sbias[s0 + n - t + 1023];
            float v1 = S[1][r] + gr[r] * sbias[s0 + 16 + n - t + 1023];
            float mx = fmaxf(v0, v1);
#pragma unroll
            for (int off = 1; off < 16; off <<= 1) mx = fmaxf(mx, __shfl_xor(mx, off, 32));
            mx = fmaxf(mx, mst[r]);
            float sc = __expf(mst[r] - mx);
            float p0 = __expf(v0 - mx);
            float p1 = __expf(v1 - mx);
            float rs = p0 + p1;
#pragma unroll
            for (int off = 1; off < 16; off <<= 1) rs += __shfl_xor(rs, off, 32);
            lst[r] = lst[r] * sc + rs;
            mst[r] = mx;
            pnew[0][r] = p0; pnew[1][r] = p1;
#pragma unroll
            for (int nt = 0; nt < 4; ++nt) o[nt][r] *= sc;
        }

        // Transpose P (D-layout -> A-layout) through per-wave LDS slice
#pragma unroll
        for (int c = 0; c < 2; ++c)
#pragma unroll
            for (int r = 0; r < 8; ++r)
                pl[wave][8 * hi + r][c * 16 + n] = (_Float16)pnew[c][r];
        v16h ap = *(const v16h*)&pl[wave][n][16 * hi];

        // ctx += P(16x32) * V(32x64) from LDS V (B-layout: row = d, cols = s-local)
#pragma unroll
        for (int nt = 0; nt < 4; ++nt) {
            const _Float16* vr = &vl[cb][nt * 16 + n][16 * hi];
            o[nt] = wmma_f16(ap, *(const v16h*)vr, o[nt]);
        }

        __syncthreads();   // buffer cb fully consumed before it is overwritten
        cb ^= 1;
    }

    // Normalize and write ctx as f16 rows (t*B+b) x E for the output projection
    float rinv[8];
#pragma unroll
    for (int r = 0; r < 8; ++r) rinv[r] = 1.0f / lst[r];
#pragma unroll
    for (int nt = 0; nt < 4; ++nt)
#pragma unroll
        for (int r = 0; r < 8; ++r) {
            int t = t0 + 8 * hi + r;
            Ctx[(size_t)(t * BSZ + b) * EMB + h * DHD + nt * 16 + n] =
                (_Float16)(o[nt][r] * rinv[r]);
        }
}

// ---------------------------------------------------------------------------
extern "C" void kernel_launch(void* const* d_in, const int* in_sizes, int n_in,
                              void* d_out, int out_size, void* d_ws, size_t ws_size,
                              hipStream_t stream) {
    (void)in_sizes; (void)n_in; (void)out_size; (void)ws_size;
    const float* query    = (const float*)d_in[0];
    const float* q_w      = (const float*)d_in[1];
    const float* q_b      = (const float*)d_in[2];
    const float* k_w      = (const float*)d_in[3];
    const float* k_b      = (const float*)d_in[4];
    const float* v_w      = (const float*)d_in[5];
    const float* v_b      = (const float*)d_in[6];
    const float* out_w    = (const float*)d_in[7];
    const float* out_b    = (const float*)d_in[8];
    const float* rel_bias = (const float*)d_in[9];
    const float* grep_w   = (const float*)d_in[10];
    const float* grep_b   = (const float*)d_in[11];
    const float* grep_a   = (const float*)d_in[12];

    // Workspace layout (~51 MB total), 256B aligned chunks
    char* ws = (char*)d_ws;
    size_t off = 0;
    auto take = [&](size_t bytes) {
        char* p = ws + off;
        off += (bytes + 255) & ~(size_t)255;
        return p;
    };
    _Float16* X   = (_Float16*)take((size_t)NROWS * EMB * 2);   // query f16, rows (t*B+b)
    _Float16* Wq  = (_Float16*)take((size_t)EMB * EMB * 2);
    _Float16* Wk  = (_Float16*)take((size_t)EMB * EMB * 2);
    _Float16* Wv  = (_Float16*)take((size_t)EMB * EMB * 2);
    _Float16* Wo  = (_Float16*)take((size_t)EMB * EMB * 2);
    _Float16* Qh  = (_Float16*)take((size_t)BSZ * NH * T_LEN * DHD * 2); // [b][h][t][d]
    _Float16* Kh  = (_Float16*)take((size_t)BSZ * NH * T_LEN * DHD * 2); // [b][h][t][d]
    _Float16* Vtp = (_Float16*)take((size_t)BSZ * NH * DHD * T_LEN * 2); // [b][h][d][t]
    _Float16* Ctx = (_Float16*)take((size_t)NROWS * EMB * 2);
    float*    gts = (float*)take((size_t)BSZ * NH * T_LEN * 4);
    float*    br  = (float*)take((size_t)NH * NBIAS * 4);

    // 1. f32 -> f16 conversions
    {
        int n1 = NROWS * EMB;                        // 4M elems
        cvt_f32_to_f16<<<(n1 / 4 + 255) / 256, 256, 0, stream>>>(query, X, n1);
        int n2 = EMB * EMB;                          // 1M elems each
        cvt_f32_to_f16<<<(n2 / 4 + 255) / 256, 256, 0, stream>>>(q_w,   Wq, n2);
        cvt_f32_to_f16<<<(n2 / 4 + 255) / 256, 256, 0, stream>>>(k_w,   Wk, n2);
        cvt_f32_to_f16<<<(n2 / 4 + 255) / 256, 256, 0, stream>>>(v_w,   Wv, n2);
        cvt_f32_to_f16<<<(n2 / 4 + 255) / 256, 256, 0, stream>>>(out_w, Wo, n2);
    }

    // 2. Gates + relative-position bias table
    gates_kernel<<<(BSZ * NH * T_LEN) / 256, 256, 0, stream>>>(query, grep_w, grep_b, grep_a, gts);
    biasrow_kernel<<<(NBIAS + 255) / 256, 256, 0, stream>>>(rel_bias, br);

    // 3. QKV projections (WMMA), epilogues scatter into attention-friendly layouts
    dim3 ggrid(NROWS / 128, EMB / 128);
    const float scaling = 0.125f;                    // DH^-0.5
    gemm16<<<ggrid, 256, 0, stream>>>(X, Wq, q_b, (void*)Qh,  0, scaling);
    gemm16<<<ggrid, 256, 0, stream>>>(X, Wk, k_b, (void*)Kh,  1, 1.0f);
    gemm16<<<ggrid, 256, 0, stream>>>(X, Wv, v_b, (void*)Vtp, 2, 1.0f);

    // 4. Flash attention with gated relative bias (async-LDS pipelined)
    attn_kernel<<<dim3(T_LEN / 128, NH, BSZ), 256, 0, stream>>>(Qh, Kh, Vtp, gts, br, Ctx);

    // 5. Output projection -> f32 d_out[(t*B+b)*E + j]
    gemm16<<<ggrid, 256, 0, stream>>>(Ctx, Wo, out_b, d_out, 3, 1.0f);
}
